// ExternalSequenceBackbonePDTModel_76699525972206
// MI455X (gfx1250) — matrix operand order repaired and checked by reading, not compiled, not run on hardware
//
#include <hip/hip_runtime.h>
#include <hip/hip_bf16.h>
#include <math.h>

typedef __attribute__((ext_vector_type(16))) _Float16 v16h;
typedef __attribute__((ext_vector_type(8)))  float    v8f;

#define Hdim 128
#define Bdim 512
#define Tdim 64
#define SWIN 2
#define GXd 512
#define NXd (GXd + 3*SWIN)   // 518
#define NYd NXd
#define MEM_ELEMS ((size_t)NXd * NYd * Hdim)   // 34,345,472 floats

__device__ __forceinline__ float sigmoidf_(float x) { return 1.0f / (1.0f + expf(-x)); }

// ---------------- zero init: memory grid + hidden state ----------------
__global__ void zero_kernel(float4* p, size_t n4) {
    size_t i = (size_t)blockIdx.x * blockDim.x + threadIdx.x;
    size_t stride = (size_t)gridDim.x * blockDim.x;
    float4 z = {0.f, 0.f, 0.f, 0.f};
    for (; i < n4; i += stride) p[i] = z;
}

// ---------------- pack weights into WMMA B-fragment lane order ----------------
// B-frag k order used here (contiguous, same as the A loads) for v_wmma_f32_16x16x32_f16 (32xK x 16xN):
//   lanes 0-15 : column N=lane,    K = kt*32 + 0..15
//   lanes 16-31: column N=lane-16, K = kt*32 + 16..31
// whhP: [kt<4][nt<32][lane<32][16 halfs]   (W_hh^T, K=128, N=512)
// woutP:[kt<8][nt<8 ][lane<32][16 halfs]   (W_out^T, K=256, N=128)
__global__ void pack_kernel(const float* __restrict__ W_hh, const float* __restrict__ W_out,
                            _Float16* __restrict__ whhP, _Float16* __restrict__ woutP) {
    int e = blockIdx.x * blockDim.x + threadIdx.x;
    if (e < 4096) {
        int kt = e >> 10, rem = e & 1023;
        int nt = rem >> 5, lane = rem & 31;
        int kb = kt * 32 + ((lane >= 16) ? 16 : 0);
        int j  = nt * 16 + (lane & 15);
        _Float16* dst = whhP + (size_t)e * 16;
        for (int i = 0; i < 16; ++i) dst[i] = (_Float16)W_hh[j * Hdim + kb + i];
    } else if (e < 4096 + 2048) {
        int e2 = e - 4096;
        int kt = e2 >> 8, rem = e2 & 255;
        int nt = rem >> 5, lane = rem & 31;
        int kb = kt * 32 + ((lane >= 16) ? 16 : 0);
        int j  = nt * 16 + (lane & 15);
        _Float16* dst = woutP + (size_t)e2 * 16;
        for (int i = 0; i < 16; ++i) dst[i] = (_Float16)W_out[j * 256 + kb + i];
    }
}

// ---------------- per-timestep compute kernel ----------------
// 32 blocks x 256 threads (8 wave32); block handles 16 batches.
__global__ __launch_bounds__(256)
void step_kernel(const float* __restrict__ ft, const float* __restrict__ W_ih,
                 const float* __restrict__ b_ih, const float* __restrict__ b_hh,
                 const float* __restrict__ b_out, const int* __restrict__ seqlen,
                 const float* __restrict__ memory, float* __restrict__ hidden,
                 float* __restrict__ od, float* __restrict__ sd, int* __restrict__ gxy,
                 const _Float16* __restrict__ whhP, const _Float16* __restrict__ woutP,
                 float* __restrict__ last, int t) {
    extern __shared__ char smem_raw[];
    float* gsum = (float*)smem_raw;            // 16*512  (gi -> gi+gh in place, except n-chunk)
    float* ghn  = gsum + 16 * 512;             // 16*128  (h_n + b_hh)
    float* qL   = ghn + 16 * 128;              // 16*128  newgate
    float* uL   = qL + 16 * 128;               // 16*128  updategate
    float* sL   = uL + 16 * 128;               // 16*128  spatialgate
    _Float16* aH = (_Float16*)(sL + 16 * 128); // 16*128  hidden f16 (A frag source)
    _Float16* aC = aH + 16 * 128;              // 16*256  concat(mix,q) f16
    int* cds = (int*)(aC + 16 * 256);          // 16*2

    const int tid  = threadIdx.x;
    const int lane = tid & 31;
    const int wave = tid >> 5;
    const int b0   = blockIdx.x * 16;

    // ---- Phase 1: hidden->f16, gi = feat @ W_ih^T + b_ih, coords ----
    for (int i = tid; i < 16 * 128; i += 256)
        aH[i] = (_Float16)hidden[(size_t)b0 * Hdim + i];
    {
        int bl = tid >> 4;
        int j0 = (tid & 15) * 32;
        const float* xt = ft + ((size_t)(b0 + bl)) * Tdim * 4 + (size_t)t * 4;
        float f0 = xt[0], f1 = xt[1];
        for (int j = j0; j < j0 + 32; ++j)
            gsum[bl * 512 + j] = f0 * W_ih[2 * j] + f1 * W_ih[2 * j + 1] + b_ih[j];
        if ((tid & 15) == 0) {
            int gx = (int)xt[2] + SWIN, gy = (int)xt[3] + SWIN;
            gx = min(max(gx, 0), NXd - 1); gy = min(max(gy, 0), NYd - 1);
            cds[bl * 2] = gx; cds[bl * 2 + 1] = gy;
            gxy[(b0 + bl) * 2] = gx; gxy[(b0 + bl) * 2 + 1] = gy;
        }
    }
    __syncthreads();

    // ---- Phase 2: gh = hidden @ W_hh^T (+biases fused in C) via WMMA ----
    {
        const int k0   = (lane >= 16) ? 16 : 0;
        const int Mrow = lane & 15;
        const int mo   = (lane >= 16) ? 8 : 0;
        const int ncol = lane & 15;
        for (int nl = 0; nl < 4; ++nl) {
            int jt  = wave * 4 + nl;        // N tile 0..31
            int j0c = jt * 16;
            bool isN = (jt >= 16) && (jt < 24);   // n-gate chunk: keep h_n separate
            float bias = b_hh[j0c + ncol];
            v8f c;
            // uniform branch (same for whole wave): avoid per-element exec masking
            if (isN) {
                for (int r = 0; r < 8; ++r) c[r] = bias;
            } else {
                for (int r = 0; r < 8; ++r) c[r] = gsum[(r + mo) * 512 + j0c + ncol] + bias;
            }
            for (int kt = 0; kt < 4; ++kt) {
                v16h a, b;
                int abase = Mrow * 128 + kt * 32 + k0;
                for (int i = 0; i < 16; ++i) a[i] = aH[abase + i];
                b = *(const v16h*)(whhP + ((size_t)(kt * 32 + jt) * 32 + lane) * 16);
                c = __builtin_amdgcn_wmma_f32_16x16x32_f16(false, a, false, b, (short)0, c, false, false);
            }
            if (isN) {
                for (int r = 0; r < 8; ++r) ghn[(r + mo) * 128 + (j0c - 256) + ncol] = c[r];
            } else {
                for (int r = 0; r < 8; ++r) gsum[(r + mo) * 512 + j0c + ncol] = c[r];
            }
        }
    }
    __syncthreads();

    // ---- Phase 3: gates ----
    for (int i = tid; i < 16 * 128; i += 256) {
        int m = i >> 7, h = i & 127;
        float r = sigmoidf_(gsum[m * 512 + h]);
        float u = sigmoidf_(gsum[m * 512 + 128 + h]);
        float s = sigmoidf_(gsum[m * 512 + 384 + h]);
        float q = tanhf(gsum[m * 512 + 256 + h] + r * ghn[m * 128 + h]);
        qL[i] = q; uL[i] = u; sL[i] = s;
        aC[m * 256 + 128 + h] = (_Float16)q;
    }
    __syncthreads();

    // ---- Phase 4: 5x5 spatial attention (one wave per 2 batches) ----
    for (int bb = 0; bb < 2; ++bb) {
        int m = wave * 2 + bb;
        int gx = cds[m * 2], gy = cds[m * 2 + 1];
        const float4 q4 = *(const float4*)(qL + m * 128 + lane * 4);
        float sc[25];
        for (int k = 0; k < 25; ++k) {
            int xi = min(max(gx + k / 5 - SWIN, 0), NXd - 1);
            int yi = min(max(gy + k % 5 - SWIN, 0), NYd - 1);
            const float4 c4 = *(const float4*)(memory + ((size_t)xi * NYd + yi) * Hdim + lane * 4);
            float p = q4.x * c4.x + q4.y * c4.y + q4.z * c4.z + q4.w * c4.w;
            for (int off = 16; off > 0; off >>= 1) p += __shfl_xor(p, off, 32);
            sc[k] = (p == 0.0f) ? -__builtin_inff() : p;
        }
        float mx = sc[0];
        for (int k = 1; k < 25; ++k) mx = fmaxf(mx, sc[k]);
        float attn[25];
        if (mx == -__builtin_inff()) {
            for (int k = 0; k < 25; ++k) attn[k] = 0.0f;
        } else {
            float ssum = 0.0f;
            for (int k = 0; k < 25; ++k) {
                float e = (sc[k] == -__builtin_inff()) ? 0.0f : expf(sc[k] - mx);
                attn[k] = e; ssum += e;
            }
            float inv = 1.0f / ssum;
            for (int k = 0; k < 25; ++k) attn[k] *= inv;
        }
        float4 acc = {0.f, 0.f, 0.f, 0.f};
        for (int k = 0; k < 25; ++k) {
            int xi = min(max(gx + k / 5 - SWIN, 0), NXd - 1);
            int yi = min(max(gy + k % 5 - SWIN, 0), NYd - 1);
            const float4 c4 = *(const float4*)(memory + ((size_t)xi * NYd + yi) * Hdim + lane * 4);
            acc.x += attn[k] * c4.x; acc.y += attn[k] * c4.y;
            acc.z += attn[k] * c4.z; acc.w += attn[k] * c4.w;
        }
        _Float16* dst = aC + m * 256 + lane * 4;
        dst[0] = (_Float16)acc.x; dst[1] = (_Float16)acc.y;
        dst[2] = (_Float16)acc.z; dst[3] = (_Float16)acc.w;
    }
    __syncthreads();

    // ---- Phase 5: atten_cs = tanh(concat @ W_out^T + b_out) via WMMA, then out ----
    {
        const int k0   = (lane >= 16) ? 16 : 0;
        const int Mrow = lane & 15;
        const int mo   = (lane >= 16) ? 8 : 0;
        const int ncol = lane & 15;
        int j0c = wave * 16;
        float bias = b_out[j0c + ncol];
        v8f c;
        for (int r = 0; r < 8; ++r) c[r] = bias;
        for (int kt = 0; kt < 8; ++kt) {
            v16h a, b;
            int abase = Mrow * 256 + kt * 32 + k0;
            for (int i = 0; i < 16; ++i) a[i] = aC[abase + i];
            b = *(const v16h*)(woutP + ((size_t)(kt * 8 + wave) * 32 + lane) * 16);
            c = __builtin_amdgcn_wmma_f32_16x16x32_f16(false, a, false, b, (short)0, c, false, false);
        }
        int hcol = j0c + ncol;
        for (int r = 0; r < 8; ++r) {
            int m = r + mo;
            int bg = b0 + m;
            float acs = tanhf(c[r]);
            float s = sL[m * 128 + hcol];
            float u = uL[m * 128 + hcol];
            float q = qL[m * 128 + hcol];
            float hid = hidden[(size_t)bg * Hdim + hcol];
            float curr = q + s * acs;
            float out = curr + u * (hid - curr);
            hidden[(size_t)bg * Hdim + hcol] = out;
            od[(size_t)bg * Hdim + hcol] = out;
            sd[(size_t)bg * Hdim + hcol] = s;
            int si = seqlen[bg]; if (si < 1) si = 1;
            if (si - 1 == t) last[(size_t)bg * Hdim + hcol] = out;
        }
    }
}

// ---------------- memory-grid scatter (deterministic last-index-wins) ----------------
__global__ __launch_bounds__(128)
void scatter_kernel(float* __restrict__ memory, const float* __restrict__ od,
                    const float* __restrict__ sd, const int* __restrict__ gxy) {
    int b = blockIdx.x;
    int h = threadIdx.x;
    __shared__ int win;
    if (h == 0) {
        int gx = gxy[2 * b], gy = gxy[2 * b + 1];
        int w = 1;
        for (int b2 = b + 1; b2 < Bdim; ++b2)
            if (gxy[2 * b2] == gx && gxy[2 * b2 + 1] == gy) { w = 0; break; }
        win = w;
    }
    __syncthreads();
    if (!win) return;
    int gx = gxy[2 * b], gy = gxy[2 * b + 1];
    size_t off = ((size_t)gx * NYd + gy) * Hdim + h;
    float s = sd[(size_t)b * Hdim + h];
    float o = od[(size_t)b * Hdim + h];
    memory[off] = s * memory[off] + (1.0f - s) * o;
}

extern "C" void kernel_launch(void* const* d_in, const int* in_sizes, int n_in,
                              void* d_out, int out_size, void* d_ws, size_t ws_size,
                              hipStream_t stream) {
    (void)in_sizes; (void)n_in; (void)out_size; (void)ws_size;
    const float* ft    = (const float*)d_in[0];
    const float* W_ih  = (const float*)d_in[1];
    const float* b_ih  = (const float*)d_in[2];
    const float* W_hh  = (const float*)d_in[3];
    const float* b_hh  = (const float*)d_in[4];
    const float* W_out = (const float*)d_in[5];
    const float* b_out = (const float*)d_in[6];
    const int*   seqln = (const int*)d_in[7];
    float* last = (float*)d_out;

    char* ws = (char*)d_ws;
    size_t off = 0;
    float* memory = (float*)(ws + off); off += MEM_ELEMS * 4;
    float* hidden = (float*)(ws + off); off += (size_t)Bdim * Hdim * 4;
    float* od     = (float*)(ws + off); off += (size_t)Bdim * Hdim * 4;
    float* sd     = (float*)(ws + off); off += (size_t)Bdim * Hdim * 4;
    int*   gxy    = (int*)(ws + off);   off += (size_t)Bdim * 2 * 4;
    _Float16* whhP  = (_Float16*)(ws + off); off += (size_t)4 * 32 * 32 * 16 * 2;
    _Float16* woutP = (_Float16*)(ws + off); off += (size_t)8 * 8 * 32 * 16 * 2;

    // memory and hidden are contiguous: zero both (hidden0 = 0, memory0 = 0)
    zero_kernel<<<2048, 256, 0, stream>>>((float4*)memory,
                                          (MEM_ELEMS + (size_t)Bdim * Hdim) / 4);
    pack_kernel<<<(4096 + 2048 + 255) / 256, 256, 0, stream>>>(W_hh, W_out, whhP, woutP);

    size_t lds = (16 * 512 + 4 * 16 * 128) * sizeof(float)   // gsum, ghn, qL, uL, sL
               + (16 * 128 + 16 * 256) * sizeof(_Float16)    // aH, aC
               + 16 * 2 * sizeof(int);                       // coords
    for (int t = 0; t < Tdim; ++t) {
        step_kernel<<<32, 256, lds, stream>>>(ft, W_ih, b_ih, b_hh, b_out, seqln,
                                              memory, hidden, od, sd, gxy,
                                              whhP, woutP, last, t);
        scatter_kernel<<<Bdim, 128, 0, stream>>>(memory, od, sd, gxy);
    }
}
